// MultiAgentLSTM_74397423501323
// MI455X (gfx1250) — matrix operand order repaired and checked by reading, not compiled, not run on hardware
//
#include <hip/hip_runtime.h>

// ---------------------------------------------------------------------------
// MultiAgentLSTM on MI455X (gfx1250): persistent-WG bf16 WMMA implementation.
// - weights staged once into LDS (bf16, padded rows), zero steady-state weight traffic
// - 64 persistent WGs (4 agents x 16 hidden-col slices), 1 grid barrier / timestep
// - dual K-bank accumulators to break WMMA->WMMA RAW hazard chains
// - bias injected via the C operand of the first WMMA (no splat moves)
// ---------------------------------------------------------------------------

typedef __attribute__((ext_vector_type(16))) __bf16 v16bf;
typedef __attribute__((ext_vector_type(8)))  __bf16 v8bf;
typedef __attribute__((ext_vector_type(8)))  float  v8f;
typedef __attribute__((ext_vector_type(8)))  unsigned short u16x8;
typedef __attribute__((ext_vector_type(4)))  unsigned short u16x4;

#define DEVI __device__ __forceinline__

constexpr int Bc = 128, Tc = 256, Nc = 4, Dc = 64, Hc = 256;
constexpr int NCHUNK  = 16;            // workgroups per agent (16 hidden cols each)
constexpr int NWG     = Nc * NCHUNK;   // 64 persistent workgroups
constexpr int NTHREAD = 256;           // 8 waves x 16 rows = 128 batch rows
constexpr int LSH = Hc + 8;            // padded LDS K-stride (K=256)
constexpr int LSD = Dc + 8;            // padded LDS K-stride (K=64)

DEVI unsigned short f2bf(float f) {     // RNE f32 -> bf16
  unsigned u = __float_as_uint(f);
  u += 0x7FFFu + ((u >> 16) & 1u);
  return (unsigned short)(u >> 16);
}

DEVI v16bf cat16(u16x8 lo, u16x8 hi) {
  v8bf l = __builtin_bit_cast(v8bf, lo);
  v8bf h = __builtin_bit_cast(v8bf, hi);
  return __builtin_shufflevector(l, h, 0,1,2,3,4,5,6,7,8,9,10,11,12,13,14,15);
}

// A fragment (16x32 bf16, M=batch rows, K contiguous in memory), per ISA layout:
// lanes 0-15: M=lane, K=kb+0..7 | kb+16..23 ; lanes 16-31: M=lane-16, K=kb+8..15 | kb+24..31
DEVI v16bf ldA(const unsigned short* __restrict__ base, int stride, int kb,
               unsigned rstMask) {
  const int lane = threadIdx.x & 31;
  const int m    = lane & 15;
  const int k0   = kb + ((lane & 16) ? 8 : 0);
  u16x8 lo = {0,0,0,0,0,0,0,0};
  u16x8 hi = {0,0,0,0,0,0,0,0};
  if (!((rstMask >> m) & 1u)) {
    const unsigned short* q = base + (size_t)m * stride + k0;
    lo = *(const u16x8*)(q);
    hi = *(const u16x8*)(q + 16);
  }
  return cat16(lo, hi);
}

// B fragment (32x16 bf16, K x N) from LDS weight slice stored as W[colLocal][k]:
// lanes 0-15: N=lane, K=kb+0..7 | kb+16..23 ; lanes 16-31: N=lane-16, K=kb+8..15 | kb+24..31
DEVI v16bf ldB(const unsigned short* __restrict__ lds, int stride, int rowBase,
               int kb) {
  const int lane = threadIdx.x & 31;
  const int c    = lane & 15;
  const int k0   = kb + ((lane & 16) ? 8 : 0);
  const unsigned short* q = lds + (rowBase + c) * stride + k0;
  u16x8 lo = *(const u16x8*)(q);
  u16x8 hi = *(const u16x8*)(q + 16);
  return cat16(lo, hi);
}

// acc[g][bank] += A x B(gate g).  Two K-banks break the WMMA RAW chain.
// INIT: first even k-step seeds bank0 from biasC[g], first odd seeds bank1 from
// zeroC — the WMMA C operand does the initialization (no v_mov splats).
template <int K, int LS, bool INIT>
DEVI void gemm4(v8f acc[4][2], const unsigned short* __restrict__ aBase,
                int aStride, const unsigned short* __restrict__ w,
                unsigned rstMask, const v8f* __restrict__ biasC,
                const v8f& zeroC) {
#pragma unroll
  for (int kb = 0; kb < K; kb += 32) {
    const int bank = (kb >> 5) & 1;
    v16bf a = ldA(aBase, aStride, kb, rstMask);
#pragma unroll
    for (int g = 0; g < 4; ++g) {
      v16bf b = ldB(w, LS, g * 16, kb);
      v8f c;
      if (INIT && kb == 0)       c = biasC[g];
      else if (INIT && kb == 32) c = zeroC;
      else                       c = acc[g][bank];
      acc[g][bank] = __builtin_amdgcn_wmma_f32_16x16x32_bf16(
          false, a, false, b, (short)0, c, false, false);
    }
  }
}

DEVI float fsig(float x)  { return 1.f / (1.f + __expf(-x)); }
DEVI float ftanh(float x) { return 1.f - 2.f / (__expf(2.f * x) + 1.f); }

// LSTM cell elementwise on one 16x16 tile; acc banks summed here.
DEVI void cell(const v8f acc[4][2], v8f& c, v8f& h, unsigned rstMask, int hiHalf) {
#pragma unroll
  for (int v = 0; v < 8; ++v) {
    float iv = fsig(acc[0][0][v] + acc[0][1][v]);
    float fv = fsig(acc[1][0][v] + acc[1][1][v]);
    float gv = ftanh(acc[2][0][v] + acc[2][1][v]);
    float ov = fsig(acc[3][0][v] + acc[3][1][v]);
    float cOld = ((rstMask >> (8 * hiHalf + v)) & 1u) ? 0.f : c[v];
    float cN = fv * cOld + iv * gv;
    c[v] = cN;
    h[v] = ov * ftanh(cN);
  }
}

DEVI void storeH(unsigned short* __restrict__ base, int m0, int j0, const v8f& h) {
  const int lane = threadIdx.x & 31;
  const int cc = lane & 15, hiHalf = lane >> 4;
#pragma unroll
  for (int v = 0; v < 8; ++v) {
    int b = m0 + 8 * hiHalf + v;
    base[(size_t)b * Hc + j0 + cc] = f2bf(h[v]);
  }
}

DEVI unsigned resetMask(const unsigned char* __restrict__ isInit, int m0, int t) {
  const int lane = threadIdx.x & 31;
  bool p = (lane < 16) && (isInit[(size_t)(m0 + lane) * Tc + t] != 0);
  return (unsigned)__builtin_amdgcn_ballot_w32(p);
}

DEVI void grid_sync(unsigned* cnt, unsigned* gen, unsigned total) {
  __threadfence();   // publish this thread's global stores (agent scope)
  __syncthreads();
  if (threadIdx.x == 0) {
    unsigned g = __hip_atomic_load(gen, __ATOMIC_RELAXED, __HIP_MEMORY_SCOPE_AGENT);
    unsigned a = __hip_atomic_fetch_add(cnt, 1u, __ATOMIC_ACQ_REL, __HIP_MEMORY_SCOPE_AGENT);
    if (a == total - 1) {
      __hip_atomic_store(cnt, 0u, __ATOMIC_RELAXED, __HIP_MEMORY_SCOPE_AGENT);
      __hip_atomic_fetch_add(gen, 1u, __ATOMIC_RELEASE, __HIP_MEMORY_SCOPE_AGENT);
    } else {
      while (__hip_atomic_load(gen, __ATOMIC_ACQUIRE, __HIP_MEMORY_SCOPE_AGENT) == g)
        __builtin_amdgcn_s_sleep(2);
    }
  }
  __syncthreads();
  __threadfence();   // acquire side: invalidate stale near caches
}

// ---------------------------------------------------------------------------
__global__ void cvt_bf16_kernel(const float* __restrict__ src,
                                unsigned short* __restrict__ dst) {
  size_t i = ((size_t)blockIdx.x * blockDim.x + threadIdx.x) * 4;
  float4 f = *reinterpret_cast<const float4*>(src + i);
  u16x4 o;
  o[0] = f2bf(f.x); o[1] = f2bf(f.y); o[2] = f2bf(f.z); o[3] = f2bf(f.w);
  *reinterpret_cast<u16x4*>(dst + i) = o;
}

// ---------------------------------------------------------------------------
__global__ __launch_bounds__(NTHREAD, 1)
void lstm_persistent(const unsigned short* __restrict__ xbf,
                     const unsigned char* __restrict__ isInit,
                     const float* __restrict__ Wih0, const float* __restrict__ Whh0,
                     const float* __restrict__ bih0, const float* __restrict__ bhh0,
                     const float* __restrict__ Wih1, const float* __restrict__ Whh1,
                     const float* __restrict__ bih1, const float* __restrict__ bhh1,
                     float* __restrict__ out,
                     unsigned short* __restrict__ h0buf,   // 2 x N x B x H bf16
                     unsigned short* __restrict__ h1buf,   // 2 x N x B x H bf16
                     unsigned* __restrict__ barBase) {     // per-agent {cnt,gen}
  __shared__ unsigned short sWih0[64 * LSD];
  __shared__ unsigned short sWhh0[64 * LSH];
  __shared__ unsigned short sWih1[64 * LSH];
  __shared__ unsigned short sWhh1[64 * LSH];
  __shared__ float sBias[2][64];

  const int n     = blockIdx.x >> 4;     // agent
  const int chunk = blockIdx.x & 15;
  const int j0    = chunk * 16;          // owned hidden cols j0..j0+15
  const int tid   = threadIdx.x;
  const int w     = tid >> 5;
  const int lane  = tid & 31;
  const int m0    = w * 16;              // owned batch rows
  const int cc    = lane & 15;
  const int hiH   = lane >> 4;

  unsigned* barCnt = barBase + (size_t)n * 64;       // 256B apart per agent
  unsigned* barGen = barCnt + 32;

  // ---- one-time: stage bf16 weight slices + fused biases into LDS ----
  for (int idx = tid; idx < 64 * Dc; idx += NTHREAD) {
    int r = idx >> 6, k = idx & 63;
    int g = r >> 4, c = r & 15;
    size_t row = (size_t)n * 4 * Hc + g * Hc + j0 + c;
    sWih0[r * LSD + k] = f2bf(Wih0[row * Dc + k]);
  }
  for (int idx = tid; idx < 64 * Hc; idx += NTHREAD) {
    int r = idx >> 8, k = idx & 255;
    int g = r >> 4, c = r & 15;
    size_t row = (size_t)n * 4 * Hc + g * Hc + j0 + c;
    sWhh0[r * LSH + k] = f2bf(Whh0[row * Hc + k]);
    sWih1[r * LSH + k] = f2bf(Wih1[row * Hc + k]);
    sWhh1[r * LSH + k] = f2bf(Whh1[row * Hc + k]);
  }
  for (int idx = tid; idx < 2 * 64; idx += NTHREAD) {
    int l = idx >> 6, r = idx & 63;
    int g = r >> 4, c = r & 15;
    size_t bi = (size_t)n * 4 * Hc + g * Hc + j0 + c;
    sBias[l][r] = (l == 0) ? (bih0[bi] + bhh0[bi]) : (bih1[bi] + bhh1[bi]);
  }
  __syncthreads();

  // persistent bias / zero registers (C-operand seeds for the first WMMA)
  v8f bias0R[4], bias1R[4];
  v8f zeroR = {0,0,0,0,0,0,0,0};
#pragma unroll
  for (int g = 0; g < 4; ++g) {
    float b0 = sBias[0][g * 16 + cc];
    float b1 = sBias[1][g * 16 + cc];
#pragma unroll
    for (int v = 0; v < 8; ++v) { bias0R[g][v] = b0; bias1R[g][v] = b1; }
  }

  const int    xStride = Tc * Nc * Dc;              // batch-row stride in xbf
  const size_t hSlab   = (size_t)Nc * Bc * Hc;      // one h double-buffer slab

  v8f c0 = {0,0,0,0,0,0,0,0}, c1 = {0,0,0,0,0,0,0,0};
  v8f h0r = {0,0,0,0,0,0,0,0}, h1r = {0,0,0,0,0,0,0,0};

  // ---- prologue: layer0 @ t=0 (h0_prev == 0 -> skip hh gemm) ----
  {
    unsigned mk = resetMask(isInit, m0, 0);
    v8f acc[4][2];
    gemm4<Dc, LSD, true>(acc, xbf + (size_t)m0 * xStride + (size_t)n * Dc,
                         xStride, sWih0, 0u, bias0R, zeroR);
    cell(acc, c0, h0r, mk, hiH);
    storeH(h0buf + (size_t)n * Bc * Hc, m0, j0, h0r);           // buffer 0
  }

  for (int t = 0; t < Tc; ++t) {
    grid_sync(barCnt, barGen, NCHUNK);
    const int rb = t & 1;          // h0(t) lives here; h1(t) written here
    const int pb = rb ^ 1;         // h1(t-1) lives here

    const unsigned short* h0cur = h0buf + (size_t)rb * hSlab + (size_t)n * Bc * Hc;

    // warm near caches for the t+2 input slice (global_prefetch_b8)
    if (t + 2 < Tc) {
      const unsigned short* pf = xbf + (size_t)(m0 + cc) * xStride +
                                 (size_t)(t + 2) * (Nc * Dc) + (size_t)n * Dc;
      __builtin_prefetch(pf, 0, 1);
    }

    // ---------- layer 1 @ t ----------
    unsigned mt = resetMask(isInit, m0, t);
    v8f a1[4][2];
    gemm4<Hc, LSH, true>(a1, h0cur + (size_t)m0 * Hc, Hc, sWih1, 0u, bias1R, zeroR);
    if (t > 0)
      gemm4<Hc, LSH, false>(a1,
                            h1buf + (size_t)pb * hSlab + ((size_t)n * Bc + m0) * Hc,
                            Hc, sWhh1, mt, bias1R, zeroR);
    cell(a1, c1, h1r, mt, hiH);
    storeH(h1buf + (size_t)rb * hSlab + (size_t)n * Bc * Hc, m0, j0, h1r);
#pragma unroll
    for (int v = 0; v < 8; ++v) {
      int b = m0 + 8 * hiH + v;
      out[(((size_t)b * Tc + t) * Nc + n) * Hc + j0 + cc] = h1r[v];
    }

    // ---------- layer 0 @ t+1 (pipelined behind the same barrier) ----------
    if (t + 1 < Tc) {
      unsigned mn = resetMask(isInit, m0, t + 1);
      v8f a0[4][2];
      gemm4<Dc, LSD, true>(a0,
                           xbf + (size_t)m0 * xStride + (size_t)(t + 1) * (Nc * Dc) +
                               (size_t)n * Dc,
                           xStride, sWih0, 0u, bias0R, zeroR);
      gemm4<Hc, LSH, false>(a0, h0cur + (size_t)m0 * Hc, Hc, sWhh0, mn, bias0R, zeroR);
      cell(a0, c0, h0r, mn, hiH);
      storeH(h0buf + (size_t)(rb ^ 1) * hSlab + (size_t)n * Bc * Hc, m0, j0, h0r);
    }
  }

  // ---- epilogue: h_n (B,N,L,H) and c_n from live registers ----
  const size_t oBase = (size_t)Bc * Tc * Nc * Hc;
  const size_t cBase = oBase + (size_t)Bc * Nc * 2 * Hc;
#pragma unroll
  for (int v = 0; v < 8; ++v) {
    int b = m0 + 8 * hiH + v;
    size_t i0 = (((size_t)b * Nc + n) * 2 + 0) * Hc + j0 + cc;
    size_t i1 = (((size_t)b * Nc + n) * 2 + 1) * Hc + j0 + cc;
    out[oBase + i0] = h0r[v];
    out[oBase + i1] = h1r[v];
    out[cBase + i0] = c0[v];
    out[cBase + i1] = c1[v];
  }
}

// ---------------------------------------------------------------------------
extern "C" void kernel_launch(void* const* d_in, const int* in_sizes, int n_in,
                              void* d_out, int out_size, void* d_ws, size_t ws_size,
                              hipStream_t stream) {
  (void)in_sizes; (void)n_in; (void)out_size; (void)ws_size;
  const float*         x      = (const float*)d_in[0];
  const unsigned char* isInit = (const unsigned char*)d_in[1];  // bool8
  const float* Wih0 = (const float*)d_in[2];
  const float* Whh0 = (const float*)d_in[3];
  const float* bih0 = (const float*)d_in[4];
  const float* bhh0 = (const float*)d_in[5];
  const float* Wih1 = (const float*)d_in[6];
  const float* Whh1 = (const float*)d_in[7];
  const float* bih1 = (const float*)d_in[8];
  const float* bhh1 = (const float*)d_in[9];
  float* out = (float*)d_out;

  char* ws = (char*)d_ws;
  unsigned*       bar  = (unsigned*)ws;                       // 1KB: 4x{cnt,gen}
  unsigned short* xbf  = (unsigned short*)(ws + 1024);
  const size_t xElems  = (size_t)Bc * Tc * Nc * Dc;           // 8,388,608
  unsigned short* h0b  = xbf + xElems;
  unsigned short* h1b  = h0b + (size_t)2 * Nc * Bc * Hc;

  hipMemsetAsync(ws, 0, 1024, stream);                        // barrier state
  cvt_bf16_kernel<<<(int)(xElems / 4 / 256), 256, 0, stream>>>(x, xbf);
  lstm_persistent<<<NWG, NTHREAD, 0, stream>>>(
      xbf, isInit, Wih0, Whh0, bih0, bhh0, Wih1, Whh1, bih1, bhh1, out,
      h0b, h1b, bar);
}
